// TensorParallelAttention_15281493639595
// MI455X (gfx1250) — matrix operand order repaired and checked by reading, not compiled, not run on hardware
//
#include <hip/hip_runtime.h>
#include <hip/hip_bf16.h>

typedef __bf16 bf16;
typedef bf16 v16bf __attribute__((ext_vector_type(16)));
typedef bf16 v8bf  __attribute__((ext_vector_type(8)));
typedef float v8f  __attribute__((ext_vector_type(8)));

#define HID   2048
#define SEQ   2048
#define NBATCH 2
#define NHEAD 16
#define HD    128
#define MTOK  4096   // B*S

__device__ __forceinline__ v8f wmma_bf16(v16bf a, v16bf b, v8f c) {
  // D = A(16x32 bf16) * B(32x16 bf16) + C(16x16 f32)
  return __builtin_amdgcn_wmma_f32_16x16x32_bf16(false, a, false, b, (short)0, c,
                                                 false, false);
}

__device__ __forceinline__ v8f zero8() {
  v8f z = {0.f, 0.f, 0.f, 0.f, 0.f, 0.f, 0.f, 0.f};
  return z;
}

// A-fragment: lane holds row m, K-elements at half*16+j (contiguous run of 16).
// p must already point at row_base + half*16.
__device__ __forceinline__ v16bf load_a_frag(const bf16* p) {
  v8bf lo = *(const v8bf*)(p);
  v8bf hi = *(const v8bf*)(p + 8);
  return __builtin_shufflevector(lo, hi, 0, 1, 2, 3, 4, 5, 6, 7,
                                 8, 9, 10, 11, 12, 13, 14, 15);
}

__device__ __forceinline__ void split2(float x, bf16& h, bf16& l) {
  h = (bf16)x;
  l = (bf16)(x - (float)h);
}

// ---------------------------------------------------------------------------
// Elementwise fp32 -> (hi, lo) bf16 split
// ---------------------------------------------------------------------------
__global__ void split_f32_kernel(const float* __restrict__ src,
                                 bf16* __restrict__ h, bf16* __restrict__ l,
                                 int n) {
  int i = blockIdx.x * blockDim.x + threadIdx.x;
  int stride = gridDim.x * blockDim.x;
  for (; i < n; i += stride) {
    float x = src[i];
    bf16 hi = (bf16)x;
    h[i] = hi;
    l[i] = (bf16)(x - (float)hi);
  }
}

// ---------------------------------------------------------------------------
// Y[M][N] = A[M][K] * W[N][K]^T with bf16x3 split WMMA.
// mode 0: write split hi/lo row-major [M][N]
// mode 1: write split hi/lo transposed [N][M]
// mode 2: write f32 row-major [M][N]
// Block: 256 threads = 8 waves (4 M-waves x 2 N-waves) -> 64M x 128N tile.
// ---------------------------------------------------------------------------
__global__ __launch_bounds__(256)
void gemm_bf16x3_kernel(const bf16* __restrict__ Ah, const bf16* __restrict__ Al,
                        const bf16* __restrict__ Bh, const bf16* __restrict__ Bl,
                        int M, int N, int K, int mode,
                        float* __restrict__ outF,
                        bf16* __restrict__ outH, bf16* __restrict__ outL) {
  const int lane = threadIdx.x & 31;
  const int wid  = threadIdx.x >> 5;
  const int ln   = lane & 15;
  const int half = lane >> 4;

  const int m0 = blockIdx.y * 64  + (wid >> 1) * 16;
  const int n0 = blockIdx.x * 128 + (wid & 1) * 64;

  const bf16* aph = Ah + (size_t)(m0 + ln) * K + half * 16;
  const bf16* apl = Al + (size_t)(m0 + ln) * K + half * 16;

  const bf16* bph[4];
  const bf16* bpl[4];
#pragma unroll
  for (int nt = 0; nt < 4; ++nt) {
    int col = n0 + nt * 16 + ln;
    bph[nt] = Bh + (size_t)col * K + half * 16;   // B-frag: contiguous K-run of 16
    bpl[nt] = Bl + (size_t)col * K + half * 16;
  }

  v8f acc[4];
#pragma unroll
  for (int nt = 0; nt < 4; ++nt) acc[nt] = zero8();

  for (int k0 = 0; k0 < K; k0 += 32) {
    v16bf a_h = load_a_frag(aph + k0);
    v16bf a_l = load_a_frag(apl + k0);
#pragma unroll
    for (int nt = 0; nt < 4; ++nt) {
      v16bf b_h = *(const v16bf*)(bph[nt] + k0);
      v16bf b_l = *(const v16bf*)(bpl[nt] + k0);
      acc[nt] = wmma_bf16(a_h, b_h, acc[nt]);
      acc[nt] = wmma_bf16(a_h, b_l, acc[nt]);
      acc[nt] = wmma_bf16(a_l, b_h, acc[nt]);
    }
    // pull the next K-slab toward the WGP (global_prefetch_b8)
    __builtin_prefetch(aph + k0 + 32, 0, 1);
    __builtin_prefetch(bph[0] + k0 + 32, 0, 1);
    __builtin_prefetch(bph[2] + k0 + 32, 0, 1);
  }

#pragma unroll
  for (int nt = 0; nt < 4; ++nt) {
#pragma unroll
    for (int r = 0; r < 8; ++r) {
      float y  = acc[nt][r];
      int grow = m0 + 8 * half + r;        // C layout: VGPR r -> row r + 8*half
      int gcol = n0 + nt * 16 + ln;        // lane -> column
      if (mode == 2) {
        outF[(size_t)grow * N + gcol] = y;
      } else {
        bf16 h, l;
        split2(y, h, l);
        size_t idx = (mode == 0) ? ((size_t)grow * N + gcol)
                                 : ((size_t)gcol * M + grow);
        outH[idx] = h;
        outL[idx] = l;
      }
    }
  }
}

// ---------------------------------------------------------------------------
// Flash attention: 1 wave per 16-query tile; block = 4 waves = 64 queries of
// one (b, h). Online softmax over key blocks of 32; P transposed C->A layout
// via per-wave LDS tile; all matmuls bf16x3 WMMA with fp32 accumulate.
// ---------------------------------------------------------------------------
__global__ __launch_bounds__(128)
void attn_kernel(const bf16* __restrict__ Qh, const bf16* __restrict__ Ql,
                 const bf16* __restrict__ Kh, const bf16* __restrict__ Kl,
                 const bf16* __restrict__ Vth, const bf16* __restrict__ Vtl,
                 const float* __restrict__ mask,
                 bf16* __restrict__ CtxH, bf16* __restrict__ CtxL) {
  const int lane = threadIdx.x & 31;
  const int wid  = threadIdx.x >> 5;
  const int ln   = lane & 15;
  const int half = lane >> 4;

  const int b  = blockIdx.x >> 4;
  const int h  = blockIdx.x & 15;
  const int q0 = blockIdx.y * 64 + wid * 16;

  __shared__ float Pl[4][16 * 32];
  float* plds = Pl[wid];

  const float scale = 0.0883883476483184f;  // 1/sqrt(128)

  // Q fragments (A layout): 4 K-steps of 32 across D=128
  const size_t qbase = (size_t)(b * SEQ + q0 + ln) * HID + h * HD;
  v16bf qfh[4], qfl[4];
#pragma unroll
  for (int kk = 0; kk < 4; ++kk) {
    qfh[kk] = load_a_frag(Qh + qbase + kk * 32 + half * 16);
    qfl[kk] = load_a_frag(Ql + qbase + kk * 32 + half * 16);
  }

  v8f acc[8];
  float mi[8], li[8];
#pragma unroll
  for (int dt = 0; dt < 8; ++dt) acc[dt] = zero8();
#pragma unroll
  for (int r = 0; r < 8; ++r) { mi[r] = -1e30f; li[r] = 0.0f; }

  for (int kb = 0; kb < SEQ / 32; ++kb) {
    const int key0 = kb * 32;

    // --- scores: two 16-key C tiles, accumulate over D=128 ---
    v8f s0 = zero8(), s1 = zero8();
    {
      const size_t kr0 = (size_t)(b * SEQ + key0 + ln) * HID + h * HD + half * 16;
      const size_t kr1 = (size_t)(b * SEQ + key0 + 16 + ln) * HID + h * HD + half * 16;
#pragma unroll
      for (int kk = 0; kk < 4; ++kk) {
        v16bf kbh0 = *(const v16bf*)(Kh + kr0 + kk * 32);
        v16bf kbl0 = *(const v16bf*)(Kl + kr0 + kk * 32);
        s0 = wmma_bf16(qfh[kk], kbh0, s0);
        s0 = wmma_bf16(qfh[kk], kbl0, s0);
        s0 = wmma_bf16(qfl[kk], kbh0, s0);
        v16bf kbh1 = *(const v16bf*)(Kh + kr1 + kk * 32);
        v16bf kbl1 = *(const v16bf*)(Kl + kr1 + kk * 32);
        s1 = wmma_bf16(qfh[kk], kbh1, s1);
        s1 = wmma_bf16(qfh[kk], kbl1, s1);
        s1 = wmma_bf16(qfl[kk], kbh1, s1);
      }
    }

    // --- online softmax update (row reductions across 16-lane halves) ---
    float p0[8], p1[8], alpha[8];
#pragma unroll
    for (int r = 0; r < 8; ++r) {
      int q = q0 + r + 8 * half;
      float v0 = s0[r] * scale + mask[(size_t)q * SEQ + key0 + ln];
      float v1 = s1[r] * scale + mask[(size_t)q * SEQ + key0 + 16 + ln];
      float tm = fmaxf(v0, v1);
      tm = fmaxf(tm, __shfl_xor(tm, 1, 32));
      tm = fmaxf(tm, __shfl_xor(tm, 2, 32));
      tm = fmaxf(tm, __shfl_xor(tm, 4, 32));
      tm = fmaxf(tm, __shfl_xor(tm, 8, 32));
      float nm = fmaxf(mi[r], tm);
      float al = __expf(mi[r] - nm);
      float e0 = __expf(v0 - nm);
      float e1 = __expf(v1 - nm);
      float ts = e0 + e1;
      ts += __shfl_xor(ts, 1, 32);
      ts += __shfl_xor(ts, 2, 32);
      ts += __shfl_xor(ts, 4, 32);
      ts += __shfl_xor(ts, 8, 32);
      li[r] = li[r] * al + ts;
      mi[r] = nm;
      alpha[r] = al;
      p0[r] = e0;
      p1[r] = e1;
    }
#pragma unroll
    for (int dt = 0; dt < 8; ++dt) {
#pragma unroll
      for (int r = 0; r < 8; ++r) acc[dt][r] *= alpha[r];
    }

    // --- P: C layout -> A layout through per-wave LDS tile ---
#pragma unroll
    for (int r = 0; r < 8; ++r) {
      plds[(r + 8 * half) * 32 + ln]      = p0[r];
      plds[(r + 8 * half) * 32 + 16 + ln] = p1[r];
    }
    __syncthreads();
    v16bf pfh, pfl;
#pragma unroll
    for (int j = 0; j < 16; ++j) {
      int col = half * 16 + j;
      float pv = plds[ln * 32 + col];
      bf16 hh, ll;
      split2(pv, hh, ll);
      pfh[j] = hh;
      pfl[j] = ll;
    }
    __syncthreads();

    // --- PV accumulate: V is transposed [hidden][token], contiguous over keys
#pragma unroll
    for (int dt = 0; dt < 8; ++dt) {
      const size_t vbase =
          (size_t)(h * HD + dt * 16 + ln) * MTOK + b * SEQ + key0 + half * 16;
      v16bf vh = *(const v16bf*)(Vth + vbase);
      v16bf vl = *(const v16bf*)(Vtl + vbase);
      acc[dt] = wmma_bf16(pfh, vh, acc[dt]);
      acc[dt] = wmma_bf16(pfh, vl, acc[dt]);
      acc[dt] = wmma_bf16(pfl, vh, acc[dt]);
    }
  }

  // --- normalize and store split context, row-major [4096][2048] ---
#pragma unroll
  for (int r = 0; r < 8; ++r) {
    float inv = 1.0f / li[r];
    int tok = b * SEQ + q0 + r + 8 * half;
#pragma unroll
    for (int dt = 0; dt < 8; ++dt) {
      float y = acc[dt][r] * inv;
      size_t idx = (size_t)tok * HID + h * HD + dt * 16 + ln;
      bf16 hh, ll;
      split2(y, hh, ll);
      CtxH[idx] = hh;
      CtxL[idx] = ll;
    }
  }
}

// ---------------------------------------------------------------------------
extern "C" void kernel_launch(void* const* d_in, const int* in_sizes, int n_in,
                              void* d_out, int out_size, void* d_ws, size_t ws_size,
                              hipStream_t stream) {
  const float* X    = (const float*)d_in[0];
  const float* mask = (const float*)d_in[1];
  const float* Wq   = (const float*)d_in[2];
  const float* Wk   = (const float*)d_in[3];
  const float* Wv   = (const float*)d_in[4];
  const float* Wo   = (const float*)d_in[5];

  const size_t MK = (size_t)MTOK * HID;   // 8,388,608 elements
  const size_t KK = (size_t)HID * HID;    // 4,194,304 elements

  char* w = (char*)d_ws;
  auto take = [&](size_t bytes) -> void* {
    void* p = (void*)w;
    w += (bytes + 255) & ~(size_t)255;
    return p;
  };

  bf16* Xh  = (bf16*)take(MK * 2);
  bf16* Xl  = (bf16*)take(MK * 2);
  bf16* WqH = (bf16*)take(KK * 2);
  bf16* WqL = (bf16*)take(KK * 2);
  bf16* WkH = (bf16*)take(KK * 2);
  bf16* WkL = (bf16*)take(KK * 2);
  bf16* WvH = (bf16*)take(KK * 2);
  bf16* WvL = (bf16*)take(KK * 2);
  bf16* WoH = (bf16*)take(KK * 2);
  bf16* WoL = (bf16*)take(KK * 2);
  bf16* Qh  = (bf16*)take(MK * 2);
  bf16* Ql  = (bf16*)take(MK * 2);
  bf16* Kh  = (bf16*)take(MK * 2);
  bf16* Kl  = (bf16*)take(MK * 2);
  bf16* VtH = (bf16*)take(MK * 2);
  bf16* VtL = (bf16*)take(MK * 2);
  // Context reuses the X split buffers (X is only read by the QKV GEMMs,
  // which complete before attn runs on the same stream).
  bf16* CtxH = Xh;
  bf16* CtxL = Xl;

  // 1) fp32 -> bf16 hi/lo splits
  split_f32_kernel<<<dim3((unsigned)((MK + 255) / 256)), 256, 0, stream>>>(X, Xh, Xl, (int)MK);
  split_f32_kernel<<<dim3((unsigned)((KK + 255) / 256)), 256, 0, stream>>>(Wq, WqH, WqL, (int)KK);
  split_f32_kernel<<<dim3((unsigned)((KK + 255) / 256)), 256, 0, stream>>>(Wk, WkH, WkL, (int)KK);
  split_f32_kernel<<<dim3((unsigned)((KK + 255) / 256)), 256, 0, stream>>>(Wv, WvH, WvL, (int)KK);
  split_f32_kernel<<<dim3((unsigned)((KK + 255) / 256)), 256, 0, stream>>>(Wo, WoH, WoL, (int)KK);

  // 2) projections: Y = X @ W^T (M=4096, N=2048, K=2048)
  dim3 ggrid(HID / 128, MTOK / 64);
  gemm_bf16x3_kernel<<<ggrid, 256, 0, stream>>>(Xh, Xl, WqH, WqL, MTOK, HID, HID,
                                                0, nullptr, Qh, Ql);
  gemm_bf16x3_kernel<<<ggrid, 256, 0, stream>>>(Xh, Xl, WkH, WkL, MTOK, HID, HID,
                                                0, nullptr, Kh, Kl);
  gemm_bf16x3_kernel<<<ggrid, 256, 0, stream>>>(Xh, Xl, WvH, WvL, MTOK, HID, HID,
                                                1, nullptr, VtH, VtL);  // V transposed

  // 3) attention -> split context (overwrites X splits, already consumed)
  attn_kernel<<<dim3(NBATCH * NHEAD, SEQ / 64), 128, 0, stream>>>(
      Qh, Ql, Kh, Kl, VtH, VtL, mask, CtxH, CtxL);

  // 4) output projection: out = Ctx @ Wo^T -> fp32 d_out
  gemm_bf16x3_kernel<<<ggrid, 256, 0, stream>>>(CtxH, CtxL, WoH, WoL, MTOK, HID,
                                                HID, 2, (float*)d_out, nullptr,
                                                nullptr);
}
